// Model_15917148799827
// MI455X (gfx1250) — matrix operand order repaired and checked by reading, not compiled, not run on hardware
//
#include <hip/hip_runtime.h>
#include <hip/hip_bf16.h>

typedef __attribute__((ext_vector_type(16))) _Float16 v16h;
typedef __attribute__((ext_vector_type(8)))  _Float16 v8h;
typedef __attribute__((ext_vector_type(8)))  float    v8f;

// ---------------------------------------------------------------------------
// Pack weights W:(KT,COUT) f32 into WMMA B-fragment order, f16, zero-padded:
//   P[((t*KC + kch)*32 + lane)*16 + h] = W[k*COUT + col]
//   col = t*16 + (lane&15),  k = kch*32 + (h<8 ? (lane>>4)*8 + h : 16 + (lane>>4)*8 + (h-8))
// ---------------------------------------------------------------------------
__global__ __launch_bounds__(256) void pack_w_kernel(
    const float* __restrict__ W, _Float16* __restrict__ P,
    int KT, int COUT, int NT, int KC)
{
    int i = blockIdx.x * blockDim.x + threadIdx.x;
    int total = NT * KC * 32 * 16;
    if (i >= total) return;
    int h    = i & 15;
    int lane = (i >> 4) & 31;
    int kch  = (i >> 9) % KC;
    int t    = (i >> 9) / KC;
    int col  = t * 16 + (lane & 15);
    int k    = kch * 32 + (h < 8 ? (lane >> 4) * 8 + h : 16 + (lane >> 4) * 8 + (h - 8));
    float w = 0.0f;
    if (k < KT && col < COUT) w = W[(size_t)k * COUT + col];
    P[i] = (_Float16)w;
}

// ---------------------------------------------------------------------------
// Fused spiral-gather + WMMA GEMM (+bias, +optional ELU)
// Activations are f16 (produced once in f16 by the previous stage);
// accumulation is f32 via v_wmma_f32_16x16x32_f16. TOUT = _Float16 or float.
// ---------------------------------------------------------------------------
template<int CIN, int COUT, bool DO_ELU, typename TOUT>
__global__ __launch_bounds__(256) void sconv_kernel(
    const _Float16* __restrict__ x, const int* __restrict__ spiral,
    const _Float16* __restrict__ W, const float* __restrict__ bias,
    TOUT* __restrict__ y, int N)
{
    constexpr int S  = 9;
    constexpr int KT = S * CIN;
    constexpr int NT = (COUT + 15) / 16;
    constexpr int KC = (KT + 31) / 32;

    const int M      = 32 * N;
    const int tilesM = M >> 4;

    __shared__ int sv[8][16][S];   // per-wave spiral vertex ids for 16 rows

    const int wave   = threadIdx.x >> 5;
    const int lane   = threadIdx.x & 31;
    const int lane15 = lane & 15;
    const int hi     = lane >> 4;

    int mtile = blockIdx.x * 8 + wave;
    const bool valid = (mtile < tilesM);
    if (!valid) mtile = tilesM - 1;          // clamp; uniform per wave
    const int rowBase = mtile << 4;

    if (lane < 16) {
        int r = rowBase + lane;
        int v = r % N;
        #pragma unroll
        for (int s = 0; s < S; ++s) sv[wave][lane][s] = spiral[v * S + s];
    }
    __syncthreads();

    const int arow = rowBase + lane15;
    const int bIdx = arow / N;
    const _Float16* __restrict__ xb = x + (size_t)bIdx * (size_t)N * CIN;
    const int* __restrict__ svp = sv[wave][lane15];
    const int kOffA = hi * 8;    // A layout: lanes 16-31 carry K+8 / K+24

    v8f acc[NT] = {};

    for (int kci = 0; kci < KC; ++kci) {
        const int kc = kci * 32;
        const bool tail = (KT % 32 != 0) && (kci == KC - 1);

        // ---- gather A fragment (16x32 f16) from spiral neighbors ----
        v16h a;
        if constexpr (CIN % 8 == 0) {
            // 8 consecutive K stay inside one neighbor row -> one b128 load
            #pragma unroll
            for (int g = 0; g < 2; ++g) {
                const int kbase = kc + kOffA + g * 16;
                v8h f = {};
                if (!tail || kbase < KT) {
                    int s  = kbase / CIN;
                    int ci = kbase - s * CIN;
                    f = *(const v8h*)(xb + (size_t)svp[s] * CIN + ci);
                }
                #pragma unroll
                for (int e = 0; e < 8; ++e) a[g * 8 + e] = f[e];
            }
        } else {
            // CIN==3 path: clamped address + value select (no exec branches)
            #pragma unroll
            for (int h = 0; h < 16; ++h) {
                int k  = kc + kOffA + (h < 8 ? h : h + 8);
                int kk = (k < KT) ? k : (KT - 1);
                int s  = kk / CIN;
                int ci = kk - s * CIN;
                _Float16 av = xb[(size_t)svp[s] * CIN + ci];
                a[h] = (k < KT) ? av : (_Float16)0.0f;
            }
        }

        // ---- B fragments: single packed 32B load per tile, then WMMA ----
        #pragma unroll
        for (int t = 0; t < NT; ++t) {
            const _Float16* wp = W + (((size_t)t * KC + kci) * 32 + lane) * 16;
            __builtin_prefetch(wp + 512, 0, 1);     // next K-chunk of this tile
            const v16h bf = *(const v16h*)wp;
            acc[t] = __builtin_amdgcn_wmma_f32_16x16x32_f16(
                false, a, false, bf, (short)0, acc[t], false, false);
        }
    }

    // epilogue: bias (+ELU), single convert per output, store.
    if (valid) {
        #pragma unroll
        for (int t = 0; t < NT; ++t) {
            const int col = t * 16 + lane15;
            if ((COUT % 16 == 0) || col < COUT) {
                const float bv = bias[col];
                #pragma unroll
                for (int j = 0; j < 8; ++j) {
                    int m = rowBase + hi * 8 + j;
                    float v = acc[t][j] + bv;
                    if (DO_ELU) v = (v > 0.0f) ? v : (__expf(v) - 1.0f);
                    y[(size_t)m * COUT + col] = (TOUT)v;
                }
            }
        }
    }
}

// ---------------------------------------------------------------------------
// Sparse pool: out[b,r,c] = sum_{j<4} vals[4r+j] * in[b,cols[4r+j],c]
// f16 in/out, f32 accumulate; vectorized 8 halves (C % 8 == 0 at all levels).
// ---------------------------------------------------------------------------
__global__ __launch_bounds__(256) void pool_kernel(
    const _Float16* __restrict__ in, _Float16* __restrict__ out,
    const int* __restrict__ cols, const float* __restrict__ vals,
    int M, int C, int Nin)
{
    const int C8 = C >> 3;
    long long i = (long long)blockIdx.x * blockDim.x + threadIdx.x;
    long long total = 32LL * M * C8;
    if (i >= total) return;
    int c8 = (int)(i % C8);
    int r  = (int)((i / C8) % M);
    int b  = (int)(i / ((long long)C8 * M));
    const _Float16* __restrict__ inb = in + (size_t)b * Nin * C;
    float acc[8] = {};
    #pragma unroll
    for (int j = 0; j < 4; ++j) {
        int   cc = cols[4 * r + j];
        float w  = vals[4 * r + j];
        const v8h v = *(const v8h*)(inb + (size_t)cc * C + c8 * 8);
        #pragma unroll
        for (int e = 0; e < 8; ++e) acc[e] += w * (float)v[e];
    }
    v8h o;
    #pragma unroll
    for (int e = 0; e < 8; ++e) o[e] = (_Float16)acc[e];
    ((v8h*)out)[i] = o;
}

// ---------------------------------------------------------------------------
// Tiled transpose: in (R,C) f32 -> out (C,R) f32
// ---------------------------------------------------------------------------
__global__ __launch_bounds__(256) void transpose32_kernel(
    const float* __restrict__ in, float* __restrict__ out, int R, int C)
{
    __shared__ float tile[32][33];
    const int r0 = blockIdx.x * 32, c0 = blockIdx.y * 32;
    for (int i = threadIdx.y; i < 32; i += 8) {
        int r = r0 + i, c = c0 + threadIdx.x;
        tile[i][threadIdx.x] = (r < R && c < C) ? in[(size_t)r * C + c] : 0.0f;
    }
    __syncthreads();
    for (int i = threadIdx.y; i < 32; i += 8) {
        int c = c0 + i, r = r0 + threadIdx.x;
        if (c < C && r < R) out[(size_t)c * R + r] = tile[threadIdx.x][i];
    }
}

// ---------------------------------------------------------------------------
// mu = sigmoid(h @ muW + mub); h f16 (32,25600), muW^T f32; wave per output.
// ---------------------------------------------------------------------------
__global__ __launch_bounds__(256) void mu_kernel(
    const _Float16* __restrict__ h, const float* __restrict__ WT,
    const float* __restrict__ bias, float* __restrict__ z, float* __restrict__ mu)
{
    int wid  = (blockIdx.x * blockDim.x + threadIdx.x) >> 5;   // 0..4095
    int lane = threadIdx.x & 31;
    if (wid >= 32 * 128) return;
    int b = wid >> 7, j = wid & 127;
    const _Float16* __restrict__ hb = h + (size_t)b * 25600;
    const float*    __restrict__ wj = WT + (size_t)j * 25600;
    float acc = 0.0f;
    for (int k = lane; k < 25600; k += 32) acc += (float)hb[k] * wj[k];
    #pragma unroll
    for (int off = 16; off > 0; off >>= 1) acc += __shfl_xor(acc, off, 32);
    if (lane == 0) {
        float v = 1.0f / (1.0f + __expf(-(acc + bias[j])));
        z[wid]  = v;
        mu[wid] = v;
    }
}

// ---------------------------------------------------------------------------
// d0 = z @ deW0 + deb0; z f32 (32,128), deW0 f32 (128,25600), out f16.
// ---------------------------------------------------------------------------
__global__ __launch_bounds__(256) void declin_kernel(
    const float* __restrict__ z, const float* __restrict__ W,
    const float* __restrict__ bias, _Float16* __restrict__ out)
{
    __shared__ float zs[128];
    const int b = blockIdx.y;
    if (threadIdx.x < 128) zs[threadIdx.x] = z[b * 128 + threadIdx.x];
    __syncthreads();
    int j = blockIdx.x * blockDim.x + threadIdx.x;
    if (j >= 25600) return;
    float acc = bias[j];
    #pragma unroll 8
    for (int k = 0; k < 128; ++k) acc += zs[k] * W[(size_t)k * 25600 + j];
    out[(size_t)b * 25600 + j] = (_Float16)acc;
}

__global__ void cvt_f16_kernel(const float* __restrict__ in,
                               _Float16* __restrict__ out, int n)
{
    int i = blockIdx.x * blockDim.x + threadIdx.x;
    if (i < n) out[i] = (_Float16)in[i];
}

// ---------------------------------------------------------------------------
extern "C" void kernel_launch(void* const* d_in, const int* in_sizes, int n_in,
                              void* d_out, int out_size, void* d_ws, size_t ws_size,
                              hipStream_t stream)
{
    (void)in_sizes; (void)n_in; (void)out_size; (void)ws_size;

    const float* x = (const float*)d_in[0];
    const int  *spiral[4], *dcols[4], *ucols[4];
    const float *dvals[4], *uvals[4], *enW[4], *enb[4];
    for (int l = 0; l < 4; ++l) {
        const int base = 1 + 9 * l;
        spiral[l] = (const int*)  d_in[base + 0];
        dcols[l]  = (const int*)  d_in[base + 2];
        dvals[l]  = (const float*)d_in[base + 3];
        ucols[l]  = (const int*)  d_in[base + 5];
        uvals[l]  = (const float*)d_in[base + 6];
        enW[l]    = (const float*)d_in[base + 7];
        enb[l]    = (const float*)d_in[base + 8];
    }
    const float* muW  = (const float*)d_in[37];
    const float* mub  = (const float*)d_in[38];
    const float* deW0 = (const float*)d_in[39];
    const float* deb0 = (const float*)d_in[40];
    const float* dW[4] = {(const float*)d_in[41], (const float*)d_in[43],
                          (const float*)d_in[45], (const float*)d_in[47]};
    const float* db[4] = {(const float*)d_in[42], (const float*)d_in[44],
                          (const float*)d_in[46], (const float*)d_in[48]};
    const float* outW = (const float*)d_in[49];
    const float* outb = (const float*)d_in[50];

    // ws layout (bytes):
    //   [0, 2MB)            packed f16 weights
    //   [2MB, 15.2MB)       muW^T f32 (128 x 25600)
    //   then: arena A (52M halves), arena B (52M halves), x16 (4.8M halves)
    char* wsb = (char*)d_ws;
    _Float16* w16  = (_Float16*)wsb;
    float*    muWT = (float*)(wsb + 2097152);
    _Float16* A    = (_Float16*)(wsb + 15204352);
    _Float16* Bb   = A + 52428800;
    _Float16* x16  = Bb + 52428800;

    // packed-weight offsets (halves): size = NT*KC*512
    enum { P_enW0 = 0,      P_enW1 = 512,    P_enW2 = 5632,   P_enW3 = 24064,
           P_dW0  = 97792,  P_dW1  = 245248, P_dW2  = 318976, P_dW3  = 337408,
           P_outW = 342016 };
    struct PackDesc { const float* src; int off, KT, COUT, NT, KC; };
    const PackDesc pk[9] = {
        {enW[0], P_enW0,   27,  16, 1,  1}, {enW[1], P_enW1,  144,  32, 2,  5},
        {enW[2], P_enW2,  288,  64, 4,  9}, {enW[3], P_enW3,  576, 128, 8, 18},
        {dW[0],  P_dW0,  1152, 128, 8, 36}, {dW[1],  P_dW1,  1152,  64, 4, 36},
        {dW[2],  P_dW2,   576,  32, 2, 18}, {dW[3],  P_dW3,   288,  16, 1,  9},
        {outW,   P_outW,  144,   3, 1,  5}};
    for (int i = 0; i < 9; ++i) {
        int n = pk[i].NT * pk[i].KC * 512;
        pack_w_kernel<<<(n + 255) / 256, 256, 0, stream>>>(
            pk[i].src, w16 + pk[i].off, pk[i].KT, pk[i].COUT, pk[i].NT, pk[i].KC);
    }
    transpose32_kernel<<<dim3(800, 4), dim3(32, 8), 0, stream>>>(muW, muWT, 25600, 128);
    cvt_f16_kernel<<<(4800000 + 255) / 256, 256, 0, stream>>>(x, x16, 4800000);

    auto scBlocks = [](int N) { return ((32 * N / 16) + 7) / 8; };
    auto poolGo = [&](const _Float16* in, _Float16* outp, const int* cols,
                      const float* vals, int M, int C, int Nin) {
        long long total = 32LL * M * (C >> 3);
        pool_kernel<<<(int)((total + 255) / 256), 256, 0, stream>>>(in, outp, cols, vals, M, C, Nin);
    };

    // ---------------- encoder ----------------
    sconv_kernel<3, 16, true, _Float16 ><<<scBlocks(50000), 256, 0, stream>>>(x16, spiral[0], w16 + P_enW0, enb[0], A, 50000);
    poolGo(A, Bb, dcols[0], dvals[0], 12500, 16, 50000);
    sconv_kernel<16, 32, true, _Float16 ><<<scBlocks(12500), 256, 0, stream>>>(Bb, spiral[1], w16 + P_enW1, enb[1], A, 12500);
    poolGo(A, Bb, dcols[1], dvals[1], 3125, 32, 12500);
    sconv_kernel<32, 64, true, _Float16 ><<<scBlocks(3125),  256, 0, stream>>>(Bb, spiral[2], w16 + P_enW2, enb[2], A, 3125);
    poolGo(A, Bb, dcols[2], dvals[2], 800, 64, 3125);
    sconv_kernel<64, 128, true, _Float16><<<scBlocks(800),   256, 0, stream>>>(Bb, spiral[3], w16 + P_enW3, enb[3], A, 800);
    poolGo(A, Bb, dcols[3], dvals[3], 200, 128, 800);      // Bb = h (32, 25600) f16

    // ---------------- latent ----------------
    float* outP = (float*)d_out;
    float* zP   = outP + 4800000;
    float* muP  = zP + 4096;
    mu_kernel<<<512, 256, 0, stream>>>(Bb, muWT, mub, zP, muP);
    declin_kernel<<<dim3(100, 32), 256, 0, stream>>>(zP, deW0, deb0, A);   // A = d0 f16

    // ---------------- decoder ----------------
    poolGo(A, Bb, ucols[3], uvals[3], 800, 128, 200);
    sconv_kernel<128, 128, true, _Float16><<<scBlocks(800),   256, 0, stream>>>(Bb, spiral[3], w16 + P_dW0, db[0], A, 800);
    poolGo(A, Bb, ucols[2], uvals[2], 3125, 128, 800);
    sconv_kernel<128, 64, true, _Float16 ><<<scBlocks(3125),  256, 0, stream>>>(Bb, spiral[2], w16 + P_dW1, db[1], A, 3125);
    poolGo(A, Bb, ucols[1], uvals[1], 12500, 64, 3125);
    sconv_kernel<64, 32, true, _Float16 ><<<scBlocks(12500), 256, 0, stream>>>(Bb, spiral[1], w16 + P_dW2, db[2], A, 12500);
    poolGo(A, Bb, ucols[0], uvals[0], 50000, 32, 12500);
    sconv_kernel<32, 16, true, _Float16 ><<<scBlocks(50000), 256, 0, stream>>>(Bb, spiral[0], w16 + P_dW3, db[3], A, 50000);
    sconv_kernel<16, 3, false, float    ><<<scBlocks(50000), 256, 0, stream>>>(A,  spiral[0], w16 + P_outW, outb, outP, 50000);
}
